// NeuronGemma4VisionAttention_74792560493262
// MI455X (gfx1250) — matrix operand order repaired and checked by reading, not compiled, not run on hardware
//
#include <hip/hip_runtime.h>

typedef __attribute__((ext_vector_type(16))) _Float16 v16h;
typedef __attribute__((ext_vector_type(8)))  _Float16 v8h;
typedef __attribute__((ext_vector_type(8)))  float    v8f;

// ---------------- constants ----------------
constexpr int Bc   = 2;
constexpr int Pc   = 4096;
constexpr int HIDc = 1152;
constexpr int NHc  = 16;
constexpr int HDc  = 72;
constexpr int DP   = 96;            // padded head dim (multiple of 32)
constexpr int Mrows = Bc * Pc;      // 8192

// ---------------- WMMA helpers ----------------
__device__ __forceinline__ v8f wmma_f16(v16h a, v16h b, v8f c) {
  // (neg_a, A, neg_b, B, c_mod, C, reuse_a, reuse_b)
  return __builtin_amdgcn_wmma_f32_16x16x32_f16(false, a, false, b, (short)0, c,
                                                false, false);
}
__device__ __forceinline__ v16h cat8(v8h lo, v8h hi) {
  return __builtin_shufflevector(lo, hi, 0,1,2,3,4,5,6,7,8,9,10,11,12,13,14,15);
}
// A-fragment (16x32 f16): per lane, K = 16*half + e (matches B) -> one
// contiguous 32B run.
__device__ __forceinline__ v16h load_a32(const _Float16* rowp, int half) {
  v8h lo = *(const v8h*)(rowp + 16 * half);
  v8h hi = *(const v8h*)(rowp + 16 * half + 8);
  return cat8(lo, hi);
}
// B-fragment (32x16 f16) when the per-lane K run is contiguous in memory:
// K = 16*half + e -> one contiguous 32B run.
__device__ __forceinline__ v16h load_b32c(const _Float16* rowp, int half) {
  v8h lo = *(const v8h*)(rowp + 16 * half);
  v8h hi = *(const v8h*)(rowp + 16 * half + 8);
  return cat8(lo, hi);
}

// ---------------- fp32 -> fp16 convert ----------------
__global__ void cvt_f32_f16(const float* __restrict__ x, _Float16* __restrict__ y, int n) {
  int i = blockIdx.x * blockDim.x + threadIdx.x;
  if (i < n) y[i] = (_Float16)x[i];
}

// ---------------- fp32 -> fp16 transpose-convert (1152x1152 weights) ----------------
// Wt[n][k] = (f16) W[k][n]; 32x32 tiles through LDS, coalesced read & write.
__global__ __launch_bounds__(256)
void transpose_cvt_kernel(const float* __restrict__ W, _Float16* __restrict__ Wt)
{
  __shared__ float tile[32][33];
  const int bx = blockIdx.x * 32;           // n base
  const int by = blockIdx.y * 32;           // k base
  const int tx = threadIdx.x & 31;
  const int ty = threadIdx.x >> 5;          // 0..7
#pragma unroll
  for (int i = 0; i < 4; ++i)
    tile[ty + 8 * i][tx] = W[(size_t)(by + ty + 8 * i) * HIDc + bx + tx];
  __syncthreads();
#pragma unroll
  for (int i = 0; i < 4; ++i)
    Wt[(size_t)(bx + ty + 8 * i) * HIDc + by + tx] = (_Float16)tile[tx][ty + 8 * i];
}

// ---------------- LDS-tiled WMMA GEMM: C(f32, Mx1152) = A(f16, Mx1152) @ Bt^T ----------
// Bt is the weight stored transposed: Bt[n][k]. Block: 256 threads = 8 waves;
// tile 128x128; wave tile 32x64 (2x4 WMMA). All LDS fragment loads are b128.
#define GBM 128
#define GBN 128
#define GBK 32
__global__ __launch_bounds__(256)
void gemm_f16f32(const _Float16* __restrict__ A, const _Float16* __restrict__ Bt,
                 float* __restrict__ C)
{
  constexpr int N = HIDc, K = HIDc;
  __shared__ __align__(16) _Float16 As[GBM][GBK + 8];   // row stride 80B
  __shared__ __align__(16) _Float16 Bs[GBN][GBK + 8];   // row stride 80B

  const int t    = threadIdx.x;
  const int lane = t & 31;
  const int wave = t >> 5;
  const int wm   = wave & 3;      // 4 M sub-tiles of 32
  const int wn   = wave >> 2;     // 2 N sub-tiles of 64
  const int half = lane >> 4;
  const int lan  = lane & 15;
  const int m0   = blockIdx.y * GBM;
  const int n0   = blockIdx.x * GBN;

  v8f acc[2][4] = {};

  const _Float16* aLd = A  + (size_t)(m0 + (t >> 1)) * K + (t & 1) * 16;
  const _Float16* bLd = Bt + (size_t)(n0 + (t >> 1)) * K + (t & 1) * 16;

  for (int kc = 0; kc < K; kc += GBK) {
    {
      int row = t >> 1, col = (t & 1) * 16;
      *(v8h*)&As[row][col]     = *(const v8h*)(aLd);
      *(v8h*)&As[row][col + 8] = *(const v8h*)(aLd + 8);
      *(v8h*)&Bs[row][col]     = *(const v8h*)(bLd);
      *(v8h*)&Bs[row][col + 8] = *(const v8h*)(bLd + 8);
      aLd += GBK;
      bLd += GBK;
    }
    __syncthreads();

    v16h af[2], bf[4];
    af[0] = load_a32(&As[wm * 32 + lan][0],      half);
    af[1] = load_a32(&As[wm * 32 + 16 + lan][0], half);
#pragma unroll
    for (int ni = 0; ni < 4; ++ni)
      bf[ni] = load_b32c(&Bs[wn * 64 + ni * 16 + lan][0], half);

#pragma unroll
    for (int mi = 0; mi < 2; ++mi)
#pragma unroll
      for (int ni = 0; ni < 4; ++ni)
        acc[mi][ni] = wmma_f16(af[mi], bf[ni], acc[mi][ni]);
    __syncthreads();
  }

#pragma unroll
  for (int mi = 0; mi < 2; ++mi) {
    float* cp = C + (size_t)(m0 + wm * 32 + mi * 16 + 8 * half) * N + n0 + wn * 64 + lan;
#pragma unroll
    for (int ni = 0; ni < 4; ++ni)
#pragma unroll
      for (int r = 0; r < 8; ++r)
        cp[r * N + ni * 16] = acc[mi][ni][r];
  }
}

// ---------------- per-head RMS-norm + 2-part RoPE + repack ----------------
// Qh/Kh: [b][h][p][96] (padded, zeros 72..95); Vt: [b][h][96][p] (transposed, rows 72..95 zero)
__global__ void rmsnorm_rope_kernel(const float* __restrict__ Qf, const float* __restrict__ Kf,
                                    const float* __restrict__ Vf,
                                    const float* __restrict__ cosb, const float* __restrict__ sinb,
                                    const float* __restrict__ qw, const float* __restrict__ kw,
                                    _Float16* __restrict__ Qh, _Float16* __restrict__ Kh,
                                    _Float16* __restrict__ Vt)
{
  int idx = blockIdx.x * blockDim.x + threadIdx.x;
  if (idx >= Mrows * NHc) return;
  int h  = idx % NHc;
  int bp = idx / NHc;
  int b  = bp / Pc;
  int p  = bp % Pc;

  const float* q = Qf + (size_t)bp * HIDc + h * HDc;
  const float* k = Kf + (size_t)bp * HIDc + h * HDc;
  const float* v = Vf + (size_t)bp * HIDc + h * HDc;
  const float* c = cosb + (size_t)bp * HDc;
  const float* s = sinb + (size_t)bp * HDc;

  float ssq = 0.f, ssk = 0.f, ssv = 0.f;
  for (int d = 0; d < HDc; ++d) {
    ssq += q[d] * q[d];
    ssk += k[d] * k[d];
    ssv += v[d] * v[d];
  }
  const float iq = rsqrtf(ssq * (1.0f / HDc) + 1e-6f);
  const float ik = rsqrtf(ssk * (1.0f / HDc) + 1e-6f);
  const float iv = rsqrtf(ssv * (1.0f / HDc) + 1e-6f);

  const size_t head = (size_t)b * NHc + h;
  _Float16* qd = Qh + (head * Pc + p) * DP;
  _Float16* kd = Kh + (head * Pc + p) * DP;
  _Float16* vd = Vt + head * DP * Pc + p;      // column p, stride Pc per dim row

  for (int d = 0; d < HDc; ++d) {
    int i  = d % 36;
    int dr = (i < 18) ? (d + 18) : (d - 18);
    float sgn = (i < 18) ? -1.0f : 1.0f;

    float qx = q[d] * iq * qw[d];
    float qr = sgn * q[dr] * iq * qw[dr];
    qd[d] = (_Float16)(qx * c[d] + qr * s[d]);

    float kx = k[d] * ik * kw[d];
    float kr = sgn * k[dr] * ik * kw[dr];
    kd[d] = (_Float16)(kx * c[d] + kr * s[d]);

    vd[(size_t)d * Pc] = (_Float16)(v[d] * iv);
  }
  for (int d = HDc; d < DP; ++d) {
    qd[d] = (_Float16)0.f;
    kd[d] = (_Float16)0.f;
    vd[(size_t)d * Pc] = (_Float16)0.f;
  }
}

// ---------------- flash attention ----------------
// Grid: (P/64, NH, B); 128 threads = 4 waves; wave owns 16 query rows; 64-key chunks.
__global__ __launch_bounds__(128)
void attn_kernel(const _Float16* __restrict__ Qh, const _Float16* __restrict__ Kh,
                 const _Float16* __restrict__ Vt, _Float16* __restrict__ AO)
{
  __shared__ __align__(16) _Float16 Ks[64][DP + 8];    // keys x dims, row 208B
  __shared__ __align__(16) _Float16 Vs[DP][64 + 8];    // dims x keys, row 144B
  __shared__ __align__(16) _Float16 Ps[4][16][64 + 8]; // per-wave P repack, row 144B

  const int b  = blockIdx.z;
  const int h  = blockIdx.y;
  const int q0 = blockIdx.x * 64;
  const int t    = threadIdx.x;
  const int lane = t & 31;
  const int w    = t >> 5;
  const int half = lane >> 4;
  const int lan  = lane & 15;

  const size_t head = (size_t)b * NHc + h;
  const _Float16* Qp = Qh + head * Pc * DP;
  const _Float16* Kp = Kh + head * Pc * DP;
  const _Float16* Vp = Vt + head * DP * Pc;

  // Q fragments: 3 K-chunks of 32 over padded 96 dims (vectorized b128 loads)
  v16h qa[3];
  {
    const _Float16* qrow = Qp + (size_t)(q0 + w * 16 + lan) * DP;
#pragma unroll
    for (int cix = 0; cix < 3; ++cix) qa[cix] = load_a32(qrow + cix * 32, half);
  }

  v8f o[5] = {};
  float mrun[8], lrun[8];
#pragma unroll
  for (int r = 0; r < 8; ++r) { mrun[r] = -1e30f; lrun[r] = 0.f; }

  for (int kb = 0; kb < Pc; kb += 64) {
    __syncthreads();
    // K tile: 64 rows x 96 dims; 2 threads/row, 48 halves each (6 x b128)
    {
      int row = t >> 1, seg = (t & 1) * 48;
      const _Float16* src = Kp + (size_t)(kb + row) * DP + seg;
#pragma unroll
      for (int i = 0; i < 6; ++i)
        *(v8h*)&Ks[row][seg + i * 8] = *(const v8h*)(src + i * 8);
    }
    // V tile (transposed source): 96 dim-rows x 64 keys; 768 8-half chunks / 128 thr
    {
#pragma unroll
      for (int i = 0; i < 6; ++i) {
        int ch = t * 6 + i;
        int row = ch >> 3, col = (ch & 7) * 8;
        *(v8h*)&Vs[row][col] = *(const v8h*)(Vp + (size_t)row * Pc + kb + col);
      }
    }
    if (kb + 64 < Pc) {   // prefetch next tiles -> global_prefetch_b8
      __builtin_prefetch(Kp + (size_t)(kb + 64 + (t >> 1)) * DP, 0, 3);
      if (t < DP) __builtin_prefetch(Vp + (size_t)t * Pc + kb + 64, 0, 3);
    }
    __syncthreads();

    // S = Q K^T : four 16-key tiles
    v8f s0 = {}, s1 = {}, s2 = {}, s3 = {};
#pragma unroll
    for (int cix = 0; cix < 3; ++cix) {
      v16h bf0 = load_b32c(&Ks[lan][cix * 32],      half);
      v16h bf1 = load_b32c(&Ks[16 + lan][cix * 32], half);
      v16h bf2 = load_b32c(&Ks[32 + lan][cix * 32], half);
      v16h bf3 = load_b32c(&Ks[48 + lan][cix * 32], half);
      s0 = wmma_f16(qa[cix], bf0, s0);
      s1 = wmma_f16(qa[cix], bf1, s1);
      s2 = wmma_f16(qa[cix], bf2, s2);
      s3 = wmma_f16(qa[cix], bf3, s3);
    }

    // online softmax per query row (element r -> row r+8*half; 16-lane reductions)
#pragma unroll
    for (int r = 0; r < 8; ++r) {
      float mx = fmaxf(fmaxf(s0[r], s1[r]), fmaxf(s2[r], s3[r]));
#pragma unroll
      for (int msk = 8; msk >= 1; msk >>= 1)
        mx = fmaxf(mx, __shfl_xor(mx, msk, 32));
      float nm = fmaxf(mrun[r], mx);
      float sc = __expf(mrun[r] - nm);
      float p0 = __expf(s0[r] - nm);
      float p1 = __expf(s1[r] - nm);
      float p2 = __expf(s2[r] - nm);
      float p3 = __expf(s3[r] - nm);
      float rs = (p0 + p1) + (p2 + p3);
#pragma unroll
      for (int msk = 8; msk >= 1; msk >>= 1)
        rs += __shfl_xor(rs, msk, 32);
      lrun[r] = lrun[r] * sc + rs;
      mrun[r] = nm;
#pragma unroll
      for (int d = 0; d < 5; ++d) o[d][r] *= sc;
      int pr = r + 8 * half;
      Ps[w][pr][lan]      = (_Float16)p0;
      Ps[w][pr][16 + lan] = (_Float16)p1;
      Ps[w][pr][32 + lan] = (_Float16)p2;
      Ps[w][pr][48 + lan] = (_Float16)p3;
    }

    // P as A-fragments (16 queries x 2x32 keys), vectorized
    v16h pa0 = load_a32(&Ps[w][lan][0],  half);
    v16h pa1 = load_a32(&Ps[w][lan][32], half);

    // O += P V : 5 dim-tiles, V B-fragments contiguous from transposed Vs
#pragma unroll
    for (int d = 0; d < 5; ++d) {
      v16h vb0 = load_b32c(&Vs[d * 16 + lan][0],  half);
      v16h vb1 = load_b32c(&Vs[d * 16 + lan][32], half);
      o[d] = wmma_f16(pa0, vb0, o[d]);
      o[d] = wmma_f16(pa1, vb1, o[d]);
    }
  }

  // epilogue: O /= l, pack to [b][p][h*72+dim] f16
#pragma unroll
  for (int r = 0; r < 8; ++r) lrun[r] = 1.0f / lrun[r];
  _Float16* aop = AO + ((size_t)b * Pc + q0 + w * 16 + 8 * half) * HIDc + h * HDc;
#pragma unroll
  for (int d = 0; d < 5; ++d) {
    int dim = d * 16 + lan;
    if (dim < HDc) {
#pragma unroll
      for (int r = 0; r < 8; ++r)
        aop[r * HIDc + dim] = (_Float16)(o[d][r] * lrun[r]);
    }
  }
}

// ---------------- host launcher ----------------
extern "C" void kernel_launch(void* const* d_in, const int* in_sizes, int n_in,
                              void* d_out, int out_size, void* d_ws, size_t ws_size,
                              hipStream_t stream)
{
  (void)in_sizes; (void)n_in; (void)out_size; (void)ws_size;
  const float* hidden = (const float*)d_in[0];
  const float* cosb   = (const float*)d_in[1];
  const float* sinb   = (const float*)d_in[2];
  const float* Wq     = (const float*)d_in[3];
  const float* Wk     = (const float*)d_in[4];
  const float* Wv     = (const float*)d_in[5];
  const float* Wo     = (const float*)d_in[6];
  const float* qw     = (const float*)d_in[7];
  const float* kw     = (const float*)d_in[8];
  float* out = (float*)d_out;

  char* ws = (char*)d_ws;
  size_t off = 0;
  auto give = [&](size_t bytes) -> char* {
    char* p = ws + off;
    off += (bytes + 255) & ~(size_t)255;
    return p;
  };

  _Float16* Xh  = (_Float16*)give((size_t)Mrows * HIDc * 2);
  _Float16* WqT = (_Float16*)give((size_t)HIDc * HIDc * 2);
  _Float16* WkT = (_Float16*)give((size_t)HIDc * HIDc * 2);
  _Float16* WvT = (_Float16*)give((size_t)HIDc * HIDc * 2);
  _Float16* WoT = (_Float16*)give((size_t)HIDc * HIDc * 2);
  float*    Qf  = (float*)give((size_t)Mrows * HIDc * 4);
  float*    Kf  = (float*)give((size_t)Mrows * HIDc * 4);
  float*    Vf  = (float*)give((size_t)Mrows * HIDc * 4);
  _Float16* Qh  = (_Float16*)give((size_t)Bc * NHc * Pc * DP * 2);
  _Float16* Kh  = (_Float16*)give((size_t)Bc * NHc * Pc * DP * 2);
  _Float16* Vt  = (_Float16*)give((size_t)Bc * NHc * Pc * DP * 2);
  _Float16* AO  = (_Float16*)give((size_t)Mrows * HIDc * 2);

  // 1) convert X; transpose-convert weights (Bt[n][k])
  {
    int nX = Mrows * HIDc;
    cvt_f32_f16<<<(nX + 255) / 256, 256, 0, stream>>>(hidden, Xh, nX);
    dim3 tg(HIDc / 32, HIDc / 32);
    transpose_cvt_kernel<<<tg, 256, 0, stream>>>(Wq, WqT);
    transpose_cvt_kernel<<<tg, 256, 0, stream>>>(Wk, WkT);
    transpose_cvt_kernel<<<tg, 256, 0, stream>>>(Wv, WvT);
    transpose_cvt_kernel<<<tg, 256, 0, stream>>>(Wo, WoT);
  }

  // 2) QKV projections
  dim3 gg(HIDc / GBN, Mrows / GBM);   // (9, 64)
  gemm_f16f32<<<gg, 256, 0, stream>>>(Xh, WqT, Qf);
  gemm_f16f32<<<gg, 256, 0, stream>>>(Xh, WkT, Kf);
  gemm_f16f32<<<gg, 256, 0, stream>>>(Xh, WvT, Vf);

  // 3) RMS-norm + RoPE + repack (V transposed)
  {
    int n = Mrows * NHc;
    rmsnorm_rope_kernel<<<(n + 255) / 256, 256, 0, stream>>>(
        Qf, Kf, Vf, cosb, sinb, qw, kw, Qh, Kh, Vt);
  }

  // 4) flash attention
  attn_kernel<<<dim3(Pc / 64, NHc, Bc), 128, 0, stream>>>(Qh, Kh, Vt, AO);

  // 5) output projection
  gemm_f16f32<<<gg, 256, 0, stream>>>(AO, WoT, out);
}
